// KFGuided_MLP_9732395892977
// MI455X (gfx1250) — matrix operand order repaired and checked by reading, not compiled, not run on hardware
//
#include <hip/hip_runtime.h>
#include <math.h>

typedef __attribute__((ext_vector_type(16))) __bf16 v16bf;
typedef __attribute__((ext_vector_type(8)))  float  v8f;

#define D_DIM 1024
#define KF    196
#define KEEP  147
#define NI0   588
#define NFIN  49
#define NCLIP 64
#define TOK   784        // 4*196 tokens per clip
#define H1    4096
#define MROWS (NCLIP*KF) // 12544

// ---------- helpers ----------
__device__ __forceinline__ float wave_reduce_sum(float v) {
#pragma unroll
  for (int off = 16; off > 0; off >>= 1) v += __shfl_xor(v, off, 32);
  return v;
}

__device__ __forceinline__ unsigned short f2bf_rne(float f) {
  unsigned int u = __float_as_uint(f);
  u += 0x7FFFu + ((u >> 16) & 1u);
  return (unsigned short)(u >> 16);
}

__device__ __forceinline__ void split_bf16(float f, unsigned short& h, unsigned short& l) {
  h = f2bf_rne(f);
  float fh = __uint_as_float(((unsigned int)h) << 16);
  l = f2bf_rne(f - fh);
}

// ---------- Kernel 0: f32 -> split-bf16 hi/lo planes (weights) ----------
__global__ __launch_bounds__(256) void split_planes_kernel(
    const float* __restrict__ src, unsigned short* __restrict__ hi,
    unsigned short* __restrict__ lo)
{
  size_t i = ((size_t)blockIdx.x * 256 + threadIdx.x) * 4;
  const float4 f = *(const float4*)(src + i);
  unsigned short h0, h1, h2, h3, g0, g1, g2, g3;
  split_bf16(f.x, h0, g0); split_bf16(f.y, h1, g1);
  split_bf16(f.z, h2, g2); split_bf16(f.w, h3, g3);
  uint2 hu, lu;
  hu.x = (unsigned)h0 | ((unsigned)h1 << 16); hu.y = (unsigned)h2 | ((unsigned)h3 << 16);
  lu.x = (unsigned)g0 | ((unsigned)g1 << 16); lu.y = (unsigned)g2 | ((unsigned)g3 << 16);
  *(uint2*)(hi + i) = hu;
  *(uint2*)(lo + i) = lu;
}

// ---------- Kernel 1: KF selection + init (one WG per clip) ----------
__global__ __launch_bounds__(256) void kf_select_kernel(
    const float* __restrict__ x, float* __restrict__ kf_sum,
    float* __restrict__ kf_count, int* __restrict__ alive,
    float* __restrict__ incr_inv)
{
  const int c = blockIdx.x;
  const float* xc = x + (size_t)c * TOK * D_DIM;
  const int tid = threadIdx.x, lane = tid & 31, wid = tid >> 5;
  __shared__ float invn[KF];
  __shared__ float S[D_DIM];
  __shared__ float red[KF];
  __shared__ int   keepf[KF];
  __shared__ int   sel[KEEP];

  for (int i = wid; i < KF; i += 8) {
    const float* row = xc + (size_t)i * D_DIM;
    float s = 0.f;
    for (int d = lane; d < D_DIM; d += 32) { float v = row[d]; s += v * v; }
    s = wave_reduce_sum(s);
    if (lane == 0) invn[i] = 1.0f / fmaxf(sqrtf(s), 1e-12f);
  }
  __syncthreads();
  // redundancy row-sum trick: sum_j kf_n_i.kf_n_j = kf_n_i . S,  S = sum_j kf_n_j
  for (int d = tid; d < D_DIM; d += 256) {
    float acc = 0.f;
    for (int i = 0; i < KF; ++i) acc += xc[(size_t)i * D_DIM + d] * invn[i];
    S[d] = acc;
  }
  __syncthreads();
  for (int i = wid; i < KF; i += 8) {
    const float* row = xc + (size_t)i * D_DIM;
    float s = 0.f;
    for (int d = lane; d < D_DIM; d += 32) s += row[d] * S[d];
    s = wave_reduce_sum(s);
    if (lane == 0) red[i] = s * invn[i] - 1.0f;
  }
  __syncthreads();
  for (int i = tid; i < KF; i += 256) {
    float ri = red[i];
    int r = 0;
    for (int j = 0; j < KF; ++j) {
      float rj = red[j];
      r += (rj < ri) || (rj == ri && j < i);
    }
    keepf[i] = (r < KEEP) ? 1 : 0;
  }
  __syncthreads();
  for (int i = tid; i < KF; i += 256) {
    if (keepf[i]) {
      int pos = 0;
      for (int j = 0; j < i; ++j) pos += keepf[j];
      sel[pos] = i;
    }
  }
  __syncthreads();
  for (int kk = wid; kk < KEEP; kk += 8) {
    const float* src = xc + (size_t)sel[kk] * D_DIM;
    float* dst = kf_sum + ((size_t)c * KEEP + kk) * D_DIM;
    for (int d = lane; d < D_DIM; d += 32) dst[d] = src[d];
    if (lane == 0) kf_count[c * KEEP + kk] = 1.0f;
  }
  for (int j = wid; j < NI0; j += 8) {
    const float* row = xc + (size_t)(KF + j) * D_DIM;
    float s = 0.f;
    for (int d = lane; d < D_DIM; d += 32) { float v = row[d]; s += v * v; }
    s = wave_reduce_sum(s);
    if (lane == 0) {
      incr_inv[c * NI0 + j] = 1.0f / fmaxf(sqrtf(s), 1e-12f);
      alive[c * NI0 + j] = j;
    }
  }
}

// ---------- Kernel 2: one merge iteration (one WG per clip) ----------
__global__ __launch_bounds__(256) void merge_iter_kernel(
    const float* __restrict__ x, float* __restrict__ kf_sum,
    float* __restrict__ kf_count,
    const int* __restrict__ alive_in, const float* __restrict__ inv_in,
    int* __restrict__ alive_out, float* __restrict__ inv_out,
    int n_cur, int n_merge)
{
  const int c = blockIdx.x;
  const float* xinc = x + (size_t)c * TOK * D_DIM + (size_t)KF * D_DIM;
  const int tid = threadIdx.x, lane = tid & 31, wid = tid >> 5;
  __shared__ float rowbuf[8][D_DIM];
  __shared__ float kfinv[KEEP];
  __shared__ float msim[NI0];
  __shared__ int   bk[NI0];
  __shared__ int   aliveL[NI0];
  __shared__ int   mflag[NI0];

  for (int j = tid; j < n_cur; j += 256) aliveL[j] = alive_in[c * NI0 + j];

  // l2norm(kf_sum/count) == kf_sum/||kf_sum||  (count cancels)
  for (int k = wid; k < KEEP; k += 8) {
    const float* row = kf_sum + ((size_t)c * KEEP + k) * D_DIM;
    float s = 0.f;
    for (int d = lane; d < D_DIM; d += 32) { float v = row[d]; s += v * v; }
    s = wave_reduce_sum(s);
    if (lane == 0) kfinv[k] = 1.0f / fmaxf(sqrtf(s), 1e-12f);
  }
  __syncthreads();

  for (int j = wid; j < n_cur; j += 8) {
    const float* row = xinc + (size_t)aliveL[j] * D_DIM;
    for (int d = lane; d < D_DIM; d += 32) rowbuf[wid][d] = row[d];
    float iinv = inv_in[c * NI0 + j];
    float best = -1e30f; int bestk = 0;
    for (int k = 0; k < KEEP; ++k) {
      const float* kr = kf_sum + ((size_t)c * KEEP + k) * D_DIM;
      float s = 0.f;
      for (int d = lane; d < D_DIM; d += 32) s += kr[d] * rowbuf[wid][d];
      s = wave_reduce_sum(s);
      float sim = s * kfinv[k] * iinv;
      if (sim > best) { best = sim; bestk = k; }  // first-max wins (argmax)
    }
    if (lane == 0) { msim[j] = best; bk[j] = bestk; }
  }
  __syncthreads();

  for (int j = tid; j < n_cur; j += 256) {
    float mj = msim[j];
    int r = 0;
    for (int l = 0; l < n_cur; ++l) {
      float ml = msim[l];
      r += (ml > mj) || (ml == mj && l < j);
    }
    mflag[j] = (r < n_merge) ? 1 : 0;
  }
  __syncthreads();

  // deterministic scatter-add: one wave owns each target kf row (no atomics)
  for (int k = wid; k < KEEP; k += 8) {
    int cnt = 0;
    for (int j = 0; j < n_cur; ++j) cnt += (mflag[j] && (bk[j] == k));
    if (cnt == 0) continue;
    float* dst = kf_sum + ((size_t)c * KEEP + k) * D_DIM;
    for (int chunk = 0; chunk < D_DIM; chunk += 32) {
      float v = dst[chunk + lane];
      for (int j = 0; j < n_cur; ++j)
        if (mflag[j] && (bk[j] == k))
          v += xinc[(size_t)aliveL[j] * D_DIM + chunk + lane];
      dst[chunk + lane] = v;
    }
    if (lane == 0) kf_count[c * KEEP + k] += (float)cnt;
  }

  for (int j = tid; j < n_cur; j += 256) {
    if (!mflag[j]) {
      int pos = 0;
      for (int l = 0; l < j; ++l) pos += !mflag[l];
      alive_out[c * NI0 + pos] = aliveL[j];
      inv_out[c * NI0 + pos]  = inv_in[c * NI0 + j];
    }
  }
}

// ---------- Kernel 3: assemble z directly as split-bf16 planes ----------
__global__ __launch_bounds__(256) void build_z_kernel(
    const float* __restrict__ x, const float* __restrict__ kf_sum,
    const float* __restrict__ kf_count, const int* __restrict__ alive_fin,
    unsigned short* __restrict__ zh, unsigned short* __restrict__ zl)
{
  const int r = blockIdx.x;   // 0..195
  const int c = blockIdx.y;   // clip
  size_t base = ((size_t)c * KF + r) * D_DIM;
  if (r < KEEP) {
    const float* src = kf_sum + ((size_t)c * KEEP + r) * D_DIM;
    float inv = 1.0f / kf_count[c * KEEP + r];
    for (int d = threadIdx.x; d < D_DIM; d += 256) {
      unsigned short h, l;
      split_bf16(src[d] * inv, h, l);
      zh[base + d] = h; zl[base + d] = l;
    }
  } else {
    int j = alive_fin[c * NI0 + (r - KEEP)];
    const float* src = x + (size_t)c * TOK * D_DIM + (size_t)(KF + j) * D_DIM;
    for (int d = threadIdx.x; d < D_DIM; d += 256) {
      unsigned short h, l;
      split_bf16(src[d], h, l);
      zh[base + d] = h; zl[base + d] = l;
    }
  }
}

// ---------- Split-bf16 WMMA GEMM over pre-converted hi/lo planes ----------
// A planes: MxK bf16 row-major; W planes: KxN bf16 row-major.
// WG tile 128x128, BK=32; 8 waves 2(M)x4(N); wave tile 64x32 = 4x2 WMMA tiles.
// D = Ah*Wh + Ah*Wl + Al*Wh per tile pair (~fp32 accuracy, f32 accumulate).
union FragBF { unsigned int u[8]; v16bf v; };

#define PITCH 40  // ushorts; 80B row pitch: 16B-aligned rows, 20-bank stride (conflict-free)

template<bool GELU, bool OUT_F32>
__global__ __launch_bounds__(256) void gemm_bf16split(
    const unsigned short* __restrict__ Ahg, const unsigned short* __restrict__ Alg,
    const unsigned short* __restrict__ Whg, const unsigned short* __restrict__ Wlg,
    const float* __restrict__ bias,
    float* __restrict__ Cf, unsigned short* __restrict__ Ch, unsigned short* __restrict__ Cl,
    int M, int N, int K)
{
  __shared__ unsigned short Ah[128][PITCH], Al[128][PITCH], Bh[128][PITCH], Bl[128][PITCH];
  const int tid = threadIdx.x, lane = tid & 31, wid = tid >> 5;
  const int m0 = blockIdx.y * 128, n0 = blockIdx.x * 128;
  const int wm = wid & 1, wn = wid >> 1;
  const int lo = lane & 15, hi = lane >> 4;

  v8f acc[4][2];
#pragma unroll
  for (int mi = 0; mi < 4; ++mi)
#pragma unroll
    for (int ni = 0; ni < 2; ++ni)
#pragma unroll
      for (int r = 0; r < 8; ++r) acc[mi][ni][r] = 0.0f;

  const int nkt = K >> 5;
  for (int kt = 0; kt < nkt; ++kt) {
    const int k0 = kt << 5;
    // ---- stage A tiles: straight uint4 copies, no conversion ----
#pragma unroll
    for (int it = 0; it < 2; ++it) {
      int q = tid + it * 256;              // 512 uint4 per plane
      int row = q >> 2, c8 = (q & 3) << 3; // 4 x (8 ushorts) per 32-wide row
      size_t g = (size_t)(m0 + row) * K + k0 + c8;
      *(uint4*)&Ah[row][c8] = *(const uint4*)(Ahg + g);
      *(uint4*)&Al[row][c8] = *(const uint4*)(Alg + g);
    }
    // ---- stage B tiles (32 x 128) transposed to [N][K] ----
#pragma unroll
    for (int it = 0; it < 2; ++it) {
      int q = tid + it * 256;
      int kk = q >> 4, c8 = (q & 15) << 3;
      size_t g = (size_t)(k0 + kk) * N + n0 + c8;
      uint4 uh = *(const uint4*)(Whg + g);
      uint4 ul = *(const uint4*)(Wlg + g);
      Bh[c8 + 0][kk] = (unsigned short)(uh.x);       Bh[c8 + 1][kk] = (unsigned short)(uh.x >> 16);
      Bh[c8 + 2][kk] = (unsigned short)(uh.y);       Bh[c8 + 3][kk] = (unsigned short)(uh.y >> 16);
      Bh[c8 + 4][kk] = (unsigned short)(uh.z);       Bh[c8 + 5][kk] = (unsigned short)(uh.z >> 16);
      Bh[c8 + 6][kk] = (unsigned short)(uh.w);       Bh[c8 + 7][kk] = (unsigned short)(uh.w >> 16);
      Bl[c8 + 0][kk] = (unsigned short)(ul.x);       Bl[c8 + 1][kk] = (unsigned short)(ul.x >> 16);
      Bl[c8 + 2][kk] = (unsigned short)(ul.y);       Bl[c8 + 3][kk] = (unsigned short)(ul.y >> 16);
      Bl[c8 + 4][kk] = (unsigned short)(ul.z);       Bl[c8 + 5][kk] = (unsigned short)(ul.z >> 16);
      Bl[c8 + 6][kk] = (unsigned short)(ul.w);       Bl[c8 + 7][kk] = (unsigned short)(ul.w >> 16);
    }
    if (kt + 1 < nkt) {
      __builtin_prefetch(Ahg + (size_t)(m0 + (tid >> 1)) * K + k0 + 32, 0, 0);
      __builtin_prefetch(Whg + (size_t)(k0 + 32 + (tid & 31)) * N + n0, 0, 0);
    }
    __syncthreads();

    // ---- B fragments: contiguous k = 2g + 16*hi (same order as A), N = lo ----
    FragBF bh[2], bl[2];
#pragma unroll
    for (int ni = 0; ni < 2; ++ni) {
      int nb = wn * 32 + ni * 16 + lo;
#pragma unroll
      for (int g = 0; g < 8; ++g) {
        int kb = 2 * g + hi * 16;
        bh[ni].u[g] = *(const unsigned int*)&Bh[nb][kb];
        bl[ni].u[g] = *(const unsigned int*)&Bl[nb][kb];
      }
    }
#pragma unroll
    for (int mi = 0; mi < 4; ++mi) {
      int mb = wm * 64 + mi * 16 + lo;
      FragBF ah, al;
#pragma unroll
      for (int g = 0; g < 8; ++g) {  // A layout, k = 2g + 16*hi (matches B)
        int ka = 2 * g + hi * 16;
        ah.u[g] = *(const unsigned int*)&Ah[mb][ka];
        al.u[g] = *(const unsigned int*)&Al[mb][ka];
      }
#pragma unroll
      for (int ni = 0; ni < 2; ++ni) {
        acc[mi][ni] = __builtin_amdgcn_wmma_f32_16x16x32_bf16(
            false, ah.v, false, bh[ni].v, (short)0, acc[mi][ni], false, false);
        acc[mi][ni] = __builtin_amdgcn_wmma_f32_16x16x32_bf16(
            false, ah.v, false, bl[ni].v, (short)0, acc[mi][ni], false, false);
        acc[mi][ni] = __builtin_amdgcn_wmma_f32_16x16x32_bf16(
            false, al.v, false, bh[ni].v, (short)0, acc[mi][ni], false, false);
      }
    }
    __syncthreads();
  }

  // epilogue: C/D layout N=lo, M=r+8*hi
#pragma unroll
  for (int mi = 0; mi < 4; ++mi)
#pragma unroll
    for (int ni = 0; ni < 2; ++ni) {
      int col = n0 + wn * 32 + ni * 16 + lo;
      float bv = bias[col];
#pragma unroll
      for (int r = 0; r < 8; ++r) {
        int row = m0 + wm * 64 + mi * 16 + hi * 8 + r;
        float v = acc[mi][ni][r] + bv;
        if (GELU) v = 0.5f * v * (1.0f + erff(v * 0.70710678118654752f));
        size_t idx = (size_t)row * N + col;
        if (OUT_F32) {
          Cf[idx] = v;
        } else {
          unsigned short h, l;
          split_bf16(v, h, l);
          Ch[idx] = h; Cl[idx] = l;
        }
      }
    }
}

// ---------- host ----------
extern "C" void kernel_launch(void* const* d_in, const int* in_sizes, int n_in,
                              void* d_out, int out_size, void* d_ws, size_t ws_size,
                              hipStream_t stream) {
  const float* x  = (const float*)d_in[0];
  const float* w1 = (const float*)d_in[1];
  const float* b1 = (const float*)d_in[2];
  const float* w2 = (const float*)d_in[3];
  const float* b2 = (const float*)d_in[4];
  float* out = (float*)d_out;

  char* wsb = (char*)d_ws;
  size_t off = 0;
  auto alloc = [&](size_t bytes) -> void* {
    off = (off + 255) & ~(size_t)255;
    void* p = wsb + off;
    off += bytes;
    return p;
  };
  float* kf_sum   = (float*)alloc((size_t)NCLIP * KEEP * D_DIM * 4);
  float* kf_count = (float*)alloc((size_t)NCLIP * KEEP * 4);
  int*   alive0   = (int*)  alloc((size_t)NCLIP * NI0 * 4);
  int*   alive1   = (int*)  alloc((size_t)NCLIP * NI0 * 4);
  float* inv0     = (float*)alloc((size_t)NCLIP * NI0 * 4);
  float* inv1     = (float*)alloc((size_t)NCLIP * NI0 * 4);
  unsigned short* wh1 = (unsigned short*)alloc((size_t)D_DIM * H1 * 2);
  unsigned short* wl1 = (unsigned short*)alloc((size_t)D_DIM * H1 * 2);
  unsigned short* wh2 = (unsigned short*)alloc((size_t)H1 * H1 * 2);
  unsigned short* wl2 = (unsigned short*)alloc((size_t)H1 * H1 * 2);
  unsigned short* zh  = (unsigned short*)alloc((size_t)MROWS * D_DIM * 2);
  unsigned short* zl  = (unsigned short*)alloc((size_t)MROWS * D_DIM * 2);
  unsigned short* hh  = (unsigned short*)alloc((size_t)MROWS * H1 * 2);
  unsigned short* hl  = (unsigned short*)alloc((size_t)MROWS * H1 * 2);

  // pre-convert weights once per launch (streaming, ~4us)
  split_planes_kernel<<<(int)((size_t)D_DIM * H1 / 1024), 256, 0, stream>>>(w1, wh1, wl1);
  split_planes_kernel<<<(int)((size_t)H1 * H1 / 1024), 256, 0, stream>>>(w2, wh2, wl2);

  kf_select_kernel<<<NCLIP, 256, 0, stream>>>(x, kf_sum, kf_count, alive0, inv0);

  int ncur = NI0, parity = 0;
  while (ncur > NFIN) {
    int third = ncur / 3, cap = ncur - NFIN;
    int nm = third < cap ? third : cap;
    if (nm <= 0) break;
    int*   ain  = parity ? alive1 : alive0;
    int*   aout = parity ? alive0 : alive1;
    float* iin  = parity ? inv1   : inv0;
    float* iout = parity ? inv0   : inv1;
    merge_iter_kernel<<<NCLIP, 256, 0, stream>>>(x, kf_sum, kf_count,
                                                 ain, iin, aout, iout, ncur, nm);
    ncur -= nm;
    parity ^= 1;
  }
  int* alive_fin = parity ? alive1 : alive0;

  build_z_kernel<<<dim3(KF, NCLIP), 256, 0, stream>>>(x, kf_sum, kf_count, alive_fin, zh, zl);

  // z(12544x1024) @ w1 + b1 -> gelu -> h (split-bf16 planes)
  gemm_bf16split<true, false><<<dim3(H1 / 128, MROWS / 128), 256, 0, stream>>>(
      zh, zl, wh1, wl1, b1, nullptr, hh, hl, MROWS, H1, D_DIM);
  // h(12544x4096) @ w2 + b2 -> out (f32)
  gemm_bf16split<false, true><<<dim3(H1 / 128, MROWS / 128), 256, 0, stream>>>(
      hh, hl, wh2, wl2, b2, out, nullptr, nullptr, MROWS, H1, H1);
}
